// MultiheadAttention_45260365365722
// MI455X (gfx1250) — compile-verified, matrix operand order checked by reading, not run on hardware
//
#include <hip/hip_runtime.h>
#include <hip/hip_bf16.h>
#include <math.h>

// ---------------------------------------------------------------------------
// Types for CDNA5 WMMA (wave32): V_WMMA_F32_16X16X32_F16
// A: 16x32 f16 (v16h per lane), B: 32x16 f16 (v16h), C/D: 16x16 f32 (v8f)
// ---------------------------------------------------------------------------
typedef __attribute__((ext_vector_type(16))) _Float16 v16h;
typedef __attribute__((ext_vector_type(8)))  _Float16 v8h;
typedef __attribute__((ext_vector_type(8)))  float    v8f;

__device__ __forceinline__ v8f wmma_f16(v16h a, v16h b, v8f c) {
    // 8 args: (neg_a, A, neg_b, B, c_mod, C, reuse_a, reuse_b)
    return __builtin_amdgcn_wmma_f32_16x16x32_f16(false, a, false, b,
                                                  (short)0, c, false, false);
}

// ISA 16-bit A 16x32 layout: lane L holds row m=L&15; halves 0..7 are
// K = 8*(L>>4)+0..7, halves 8..15 are K = 16+8*(L>>4)+0..7.  B mirrors this
// with lane = column n.  Both fragments = two contiguous 16-byte loads.
__device__ __forceinline__ v16h load_frag16(const _Float16* rowp, int lane) {
    int base = (lane >> 4) << 3;                 // 0 or 8
    v8h lo = *(const v8h*)(rowp + base);
    v8h hi = *(const v8h*)(rowp + 16 + base);
    return __builtin_shufflevector(lo, hi, 0,1,2,3,4,5,6,7,
                                           8,9,10,11,12,13,14,15);
}

// xor-lane float max via ds_swizzle (group-of-32 xor mode): j = i ^ mask
#define XOR_MAX(v, imm)                                                        \
    (v) = fmaxf((v), __int_as_float(__builtin_amdgcn_ds_swizzle(               \
                         __float_as_int(v), (imm))))

// ---------------------------------------------------------------------------
// fp32 -> f16 convert
// ---------------------------------------------------------------------------
__global__ void f32_to_f16_kernel(const float* __restrict__ src,
                                  _Float16* __restrict__ dst, int n) {
    int i = blockIdx.x * blockDim.x + threadIdx.x;
    if (i < n) dst[i] = (_Float16)src[i];
}

// ---------------------------------------------------------------------------
// out[MxN] = A[MxK] @ W[NxK]^T   (f16 in, fp32 accumulate)
// One wave computes a 32x64 strip (2x4 tiles): per k-step 2 A-frags + 4
// B-frags feed 8 WMMAs.  Grid supplies exactly (M/32)*(N/64) waves.
// ---------------------------------------------------------------------------
template <bool OUT_F32>
__global__ void wmma_gemm_kernel(const _Float16* __restrict__ A,
                                 const _Float16* __restrict__ W,
                                 void* __restrict__ outp,
                                 int M, int N, int K) {
    int lane = threadIdx.x & 31;
    int wv   = blockIdx.x * (blockDim.x >> 5) + (threadIdx.x >> 5);
    int ngrp = N >> 6;                 // groups of 4 tiles (64 cols)
    int ti = wv / ngrp;
    int tg = wv % ngrp;
    int lm = lane & 15;

    const _Float16* arow0 = A + (size_t)(ti * 32 + lm) * K;
    const _Float16* arow1 = A + (size_t)(ti * 32 + 16 + lm) * K;
    const _Float16* brow0 = W + (size_t)(tg * 64 + lm) * K;

    v8f acc00 = {}, acc01 = {}, acc02 = {}, acc03 = {};
    v8f acc10 = {}, acc11 = {}, acc12 = {}, acc13 = {};
    for (int k0 = 0; k0 < K; k0 += 32) {
        v16h a0 = load_frag16(arow0 + k0, lane);
        v16h a1 = load_frag16(arow1 + k0, lane);
        v16h b;
        b = load_frag16(brow0 + k0, lane);
        acc00 = wmma_f16(a0, b, acc00);
        acc10 = wmma_f16(a1, b, acc10);
        b = load_frag16(brow0 + (size_t)16 * K + k0, lane);
        acc01 = wmma_f16(a0, b, acc01);
        acc11 = wmma_f16(a1, b, acc11);
        b = load_frag16(brow0 + (size_t)32 * K + k0, lane);
        acc02 = wmma_f16(a0, b, acc02);
        acc12 = wmma_f16(a1, b, acc12);
        b = load_frag16(brow0 + (size_t)48 * K + k0, lane);
        acc03 = wmma_f16(a0, b, acc03);
        acc13 = wmma_f16(a1, b, acc13);
    }

    int hi8 = (lane >> 4) << 3;
    int col = tg * 64 + lm;
    if (OUT_F32) {
        float* out = (float*)outp;
#pragma unroll
        for (int r = 0; r < 8; ++r) {
            size_t rb0 = (size_t)(ti * 32 + r + hi8) * N + col;
            size_t rb1 = (size_t)(ti * 32 + 16 + r + hi8) * N + col;
            out[rb0 + 0]  = acc00[r];
            out[rb0 + 16] = acc01[r];
            out[rb0 + 32] = acc02[r];
            out[rb0 + 48] = acc03[r];
            out[rb1 + 0]  = acc10[r];
            out[rb1 + 16] = acc11[r];
            out[rb1 + 32] = acc12[r];
            out[rb1 + 48] = acc13[r];
        }
    } else {
        _Float16* out = (_Float16*)outp;
#pragma unroll
        for (int r = 0; r < 8; ++r) {
            size_t rb0 = (size_t)(ti * 32 + r + hi8) * N + col;
            size_t rb1 = (size_t)(ti * 32 + 16 + r + hi8) * N + col;
            out[rb0 + 0]  = (_Float16)acc00[r];
            out[rb0 + 16] = (_Float16)acc01[r];
            out[rb0 + 32] = (_Float16)acc02[r];
            out[rb0 + 48] = (_Float16)acc03[r];
            out[rb1 + 0]  = (_Float16)acc10[r];
            out[rb1 + 16] = (_Float16)acc11[r];
            out[rb1 + 32] = (_Float16)acc12[r];
            out[rb1 + 48] = (_Float16)acc13[r];
        }
    }
}

// ---------------------------------------------------------------------------
// RoPE on q/k + rearrange.  Q is pre-scaled by 1/sqrt(64) so the attention
// kernel's WMMA produces already-scaled logits.
//   Qh/Kh : f16 [bh][2048][64]   (rows along head-dim)
//   Vt    : f16 [bh][64][2048]   (rows along keys)
// ---------------------------------------------------------------------------
__global__ void rope_rearrange_kernel(const _Float16* __restrict__ qf,
                                      const _Float16* __restrict__ kf,
                                      const _Float16* __restrict__ vf,
                                      const float* __restrict__ cosp,
                                      const float* __restrict__ sinp,
                                      _Float16* __restrict__ Qh,
                                      _Float16* __restrict__ Kh,
                                      _Float16* __restrict__ Vt) {
    int idx = blockIdx.x * blockDim.x + threadIdx.x;   // 0 .. 4096*384-1
    int m  = idx / 384;
    int pc = idx % 384;
    int c  = pc << 1;
    int b  = m >> 11;
    int t  = m & 2047;
    int h  = c >> 6;
    int d  = c & 63;

    size_t src = (size_t)m * 768 + c;
    float q0 = (float)qf[src], q1 = (float)qf[src + 1];
    float k0 = (float)kf[src], k1 = (float)kf[src + 1];
    float v0 = (float)vf[src], v1 = (float)vf[src + 1];

    float cs0 = cosp[t * 64 + d],     sn0 = sinp[t * 64 + d];
    float cs1 = cosp[t * 64 + d + 1], sn1 = sinp[t * 64 + d + 1];

    const float QSCALE = 0.125f;   // 1/sqrt(head_dim)
    float qr0 = (q0 * cs0 - q1 * sn0) * QSCALE;
    float qr1 = (q1 * cs1 + q0 * sn1) * QSCALE;
    float kr0 = k0 * cs0 - k1 * sn0;
    float kr1 = k1 * cs1 + k0 * sn1;

    int bh = b * 12 + h;
    size_t qdst = ((size_t)bh * 2048 + t) * 64 + d;
    Qh[qdst]     = (_Float16)qr0;
    Qh[qdst + 1] = (_Float16)qr1;
    Kh[qdst]     = (_Float16)kr0;
    Kh[qdst + 1] = (_Float16)kr1;

    size_t vdst = ((size_t)bh * 64 + d) * 2048 + t;
    Vt[vdst]        = (_Float16)v0;
    Vt[vdst + 2048] = (_Float16)v1;
}

// ---------------------------------------------------------------------------
// Causal flash attention, one wave per (bh, 16-query tile), 32 keys/step.
//  - 4 WMMAs for the 16x32 score block
//  - tile-wide running max (any per-row constant is softmax-invariant);
//    per-lane v_max3 chain + 5 ds_swizzle xor hops
//  - raw v_exp_f32 via __builtin_amdgcn_exp2f (flushes -huge to 0 for mask)
//  - alpha-rescale only when the running max actually moves (wave-uniform)
//  - row sums via a 5th WMMA against an all-ones B fragment
//  - P restaged through 1KB LDS/wave, then 5 WMMAs (4 V-tiles + ones)
// ---------------------------------------------------------------------------
__global__ void attn_kernel(const _Float16* __restrict__ Qh,
                            const _Float16* __restrict__ Kh,
                            const _Float16* __restrict__ Vt,
                            _Float16* __restrict__ Yh) {
    __shared__ alignas(16) _Float16 plds_all[4][16 * 32];

    int lane  = threadIdx.x & 31;
    int wslot = threadIdx.x >> 5;
    _Float16* plds = plds_all[wslot];

    int wv = blockIdx.x * 4 + wslot;
    int bh = wv >> 7;
    int i0 = (wv & 127) << 4;
    int b  = bh / 12;
    int h  = bh % 12;

    const _Float16* Qp = Qh + (size_t)bh * 2048 * 64;
    const _Float16* Kp = Kh + (size_t)bh * 2048 * 64;
    const _Float16* Vp = Vt + (size_t)bh * 64 * 2048;

    int lm  = lane & 15;
    int hi8 = (lane >> 4) << 3;

    v16h qa0 = load_frag16(Qp + (size_t)(i0 + lm) * 64 + 0,  lane);
    v16h qa1 = load_frag16(Qp + (size_t)(i0 + lm) * 64 + 32, lane);

    v16h ones;
#pragma unroll
    for (int i = 0; i < 16; ++i) ones[i] = (_Float16)1.0f;

    float m_run = -3.0e38f;
    v8f o0 = {}, o1 = {}, o2 = {}, o3 = {}, o4 = {};   // o4 = row sums (l)

    const float LOG2E = 1.4426950408889634f;
    const float NEG   = -3.0e38f;

    for (int j0 = 0; j0 < i0 + 16; j0 += 32) {
        v8f s0 = {}, s1 = {};
        {
            v16h kb;
            kb = load_frag16(Kp + (size_t)(j0 + lm) * 64 + 0,  lane);
            s0 = wmma_f16(qa0, kb, s0);
            kb = load_frag16(Kp + (size_t)(j0 + lm) * 64 + 32, lane);
            s0 = wmma_f16(qa1, kb, s0);
            kb = load_frag16(Kp + (size_t)(j0 + 16 + lm) * 64 + 0,  lane);
            s1 = wmma_f16(qa0, kb, s1);
            kb = load_frag16(Kp + (size_t)(j0 + 16 + lm) * 64 + 32, lane);
            s1 = wmma_f16(qa1, kb, s1);
        }

        // causal mask + per-lane max over this lane's 16 scores
        float a0v[8], a1v[8];
        float pm = NEG;
#pragma unroll
        for (int r = 0; r < 8; ++r) {
            int row = i0 + r + hi8;
            int col = j0 + lm;
            float a0 = s0[r]; if (col > row)      a0 = NEG;
            float a1 = s1[r]; if (col + 16 > row) a1 = NEG;
            a0v[r] = a0; a1v[r] = a1;
            pm = fmaxf(pm, fmaxf(a0, a1));
        }
        // wave-wide (tile-wide) max: xor hops 1,2,4,8,16
        XOR_MAX(pm, 0x041f);
        XOR_MAX(pm, 0x081f);
        XOR_MAX(pm, 0x101f);
        XOR_MAX(pm, 0x201f);
        XOR_MAX(pm, 0x401f);

        // Rescale only when the running max moves (wave-uniform condition;
        // VALU-only inside, WMMA stays unconditional with EXEC all-ones).
        if (pm > m_run) {
            float mn    = pm;
            float alpha = __builtin_amdgcn_exp2f((m_run - mn) * LOG2E);
            m_run = mn;
#pragma unroll
            for (int r = 0; r < 8; ++r) {
                o0[r] *= alpha; o1[r] *= alpha; o2[r] *= alpha;
                o3[r] *= alpha; o4[r] *= alpha;
            }
        }
        float mn = m_run;

#pragma unroll
        for (int r = 0; r < 8; ++r) {
            float p0 = __builtin_amdgcn_exp2f((a0v[r] - mn) * LOG2E);
            float p1 = __builtin_amdgcn_exp2f((a1v[r] - mn) * LOG2E);
            int prow = r + hi8;
            plds[prow * 32 + lm]      = (_Float16)p0;
            plds[prow * 32 + lm + 16] = (_Float16)p1;
        }

        // Same-wave DS ordering: drain LDS stores before fragment reload.
        asm volatile("s_wait_dscnt 0" ::: "memory");

        v16h pa = load_frag16(plds + lm * 32, lane);
        v16h vb;
        vb = load_frag16(Vp + (size_t)(0  + lm) * 2048 + j0, lane);
        o0 = wmma_f16(pa, vb, o0);
        vb = load_frag16(Vp + (size_t)(16 + lm) * 2048 + j0, lane);
        o1 = wmma_f16(pa, vb, o1);
        vb = load_frag16(Vp + (size_t)(32 + lm) * 2048 + j0, lane);
        o2 = wmma_f16(pa, vb, o2);
        vb = load_frag16(Vp + (size_t)(48 + lm) * 2048 + j0, lane);
        o3 = wmma_f16(pa, vb, o3);
        o4 = wmma_f16(pa, ones, o4);          // row sums, same rescaling
    }

#pragma unroll
    for (int r = 0; r < 8; ++r) {
        float inv = 1.0f / o4[r];             // l for this row
        int t = i0 + r + hi8;
        size_t base = ((size_t)(b * 2048 + t)) * 768 + (size_t)h * 64 + lm;
        Yh[base + 0]  = (_Float16)(o0[r] * inv);
        Yh[base + 16] = (_Float16)(o1[r] * inv);
        Yh[base + 32] = (_Float16)(o2[r] * inv);
        Yh[base + 48] = (_Float16)(o3[r] * inv);
    }
}

// ---------------------------------------------------------------------------
// Host-side orchestration
// ---------------------------------------------------------------------------
extern "C" void kernel_launch(void* const* d_in, const int* in_sizes, int n_in,
                              void* d_out, int out_size, void* d_ws, size_t ws_size,
                              hipStream_t stream) {
    const float* x  = (const float*)d_in[0];
    const float* wq = (const float*)d_in[1];
    const float* wk = (const float*)d_in[2];
    const float* wv = (const float*)d_in[3];
    const float* wo = (const float*)d_in[4];
    const float* cs = (const float*)d_in[5];
    const float* sn = (const float*)d_in[6];

    const int M = 2 * 2048;   // tokens (B*T)
    const int C = 768;

    size_t off = 0;
    auto bump = [&](size_t bytes) -> char* {
        char* p = (char*)d_ws + off;
        off += (bytes + 255) & ~(size_t)255;
        return p;
    };
    _Float16* xh  = (_Float16*)bump((size_t)M * C * 2);
    _Float16* wqh = (_Float16*)bump((size_t)C * C * 2);
    _Float16* wkh = (_Float16*)bump((size_t)C * C * 2);
    _Float16* wvh = (_Float16*)bump((size_t)C * C * 2);
    _Float16* woh = (_Float16*)bump((size_t)C * C * 2);
    _Float16* qf  = (_Float16*)bump((size_t)M * C * 2);
    _Float16* kf  = (_Float16*)bump((size_t)M * C * 2);
    _Float16* vf  = (_Float16*)bump((size_t)M * C * 2);
    _Float16* Qh  = (_Float16*)bump((size_t)M * C * 2);
    _Float16* Kh  = (_Float16*)bump((size_t)M * C * 2);
    _Float16* Vt  = (_Float16*)bump((size_t)M * C * 2);
    _Float16* Yh  = (_Float16*)bump((size_t)M * C * 2);

    // 1) fp32 -> f16 conversions
    {
        int n = M * C;
        f32_to_f16_kernel<<<(n + 255) / 256, 256, 0, stream>>>(x, xh, n);
        int nw = C * C;
        int blk = (nw + 255) / 256;
        f32_to_f16_kernel<<<blk, 256, 0, stream>>>(wq, wqh, nw);
        f32_to_f16_kernel<<<blk, 256, 0, stream>>>(wk, wkh, nw);
        f32_to_f16_kernel<<<blk, 256, 0, stream>>>(wv, wvh, nw);
        f32_to_f16_kernel<<<blk, 256, 0, stream>>>(wo, woh, nw);
    }

    // 2) Q/K/V projections (x @ W^T), f16 outputs.  32x64 strip per wave.
    {
        int waves  = (M / 32) * (C / 64);   // 128 * 12 = 1536
        int blocks = waves / 4;             // 4 waves (128 threads) per block
        wmma_gemm_kernel<false><<<blocks, 128, 0, stream>>>(xh, wqh, qf, M, C, C);
        wmma_gemm_kernel<false><<<blocks, 128, 0, stream>>>(xh, wkh, kf, M, C, C);
        wmma_gemm_kernel<false><<<blocks, 128, 0, stream>>>(xh, wvh, vf, M, C, C);
    }

    // 3) RoPE (+1/sqrt(D) folded into Q) + rearrange to per-head layouts
    {
        int pairs = M * C / 2;
        rope_rearrange_kernel<<<pairs / 256, 256, 0, stream>>>(
            qf, kf, vf, cs, sn, Qh, Kh, Vt);
    }

    // 4) Causal flash attention
    {
        int waves = 24 * 128;
        attn_kernel<<<waves / 4, 128, 0, stream>>>(Qh, Kh, Vt, Yh);
    }

    // 5) Output projection (y @ wo^T), fp32 output
    {
        int waves = (M / 32) * (C / 64);
        wmma_gemm_kernel<true><<<waves / 4, 128, 0, stream>>>(
            Yh, woh, (float*)d_out, M, C, C);
    }
}
